// BiGRU_8727373546273
// MI455X (gfx1250) — compile-verified, matrix operand order checked by reading, not run on hardware
//
#include <hip/hip_runtime.h>

// ---------------- problem constants ----------------
#define B_  64
#define T_  512
#define D_  256
#define U_  512
#define NG  1536          // 3*U
#define OUTW 1024         // 2*U

// ---------------- launch configuration -------------
#define NWG 64
#define WAVES_PER_WG 4
#define NWAVES (NWG * WAVES_PER_WG)   // 256 waves

// phase A: [B,768] x [768,1536] per dir; 2dir x 4mtile row-groups, 96 ntiles each.
//          32 waves per row-group; each wave owns ntiles {wig, wig+32, wig+64}
//          == one u-gate tile, one r-gate tile, one candidate tile. K-tiles = 24.
// phase B: [B,512] x [512,512] per dir -> 256 tiles, exactly 1 per wave. K-tiles = 16.
#define KT_A    24
#define KT_B    16

typedef __attribute__((ext_vector_type(16))) __bf16 v16bf;
typedef __attribute__((ext_vector_type(8)))  __bf16 v8bf;
typedef __attribute__((ext_vector_type(8)))  float  v8f;

__device__ __forceinline__ unsigned short f2bf(float f) {
  unsigned u = __builtin_bit_cast(unsigned, f);
  u += 0x7FFFu + ((u >> 16) & 1u);        // round-to-nearest-even
  return (unsigned short)(u >> 16);
}

__device__ __forceinline__ float sigm(float x) {
  return 1.0f / (1.0f + __expf(-x));
}
__device__ __forceinline__ float ftanh(float x) {
  x = fminf(15.0f, fmaxf(-15.0f, x));
  float e = __expf(2.0f * x);
  return (e - 1.0f) / (e + 1.0f);
}

// monotonic grid barrier (counter zeroed by init kernel each launch)
__device__ __forceinline__ void grid_sync(unsigned* cnt, unsigned& target) {
  __syncthreads();
  if (threadIdx.x == 0) {
    __threadfence();                       // release prior stores device-wide
    atomicAdd(cnt, 1u);
    target += NWG;
    while (__hip_atomic_load(cnt, __ATOMIC_ACQUIRE, __HIP_MEMORY_SCOPE_AGENT) < target) {
      __builtin_amdgcn_s_sleep(1);
    }
  }
  __syncthreads();
}

// -------- build v16bf A-operand from two contiguous 8-half chunks --------
__device__ __forceinline__ v16bf load_a(const unsigned short* src, int k0) {
  v8bf lo = *(const v8bf*)(src + k0);
  v8bf hi = *(const v8bf*)(src + k0 + 16);
  v16bf a;
#pragma unroll
  for (int i = 0; i < 8; ++i) { a[i] = lo[i]; a[i + 8] = hi[i]; }
  return a;
}

__device__ __forceinline__ v8f wmma_bf16(v16bf a, v16bf b, v8f c) {
  return __builtin_amdgcn_wmma_f32_16x16x32_bf16(false, a, false, b, (short)0, c,
                                                 false, false);
}

// ============================ prep kernels ============================

__global__ void k_init(float* h32, unsigned short* hb, unsigned* cnt) {
  size_t i = (size_t)blockIdx.x * blockDim.x + threadIdx.x;
  if (i == 0) *cnt = 0u;
  if (i < (size_t)2 * B_ * U_) { h32[i] = 0.0f; hb[i] = 0; }
}

// x [B,T,D] f32 -> xb [T,B,D] bf16
__global__ void k_convert_x(const float* __restrict__ x, unsigned short* __restrict__ xb) {
  size_t i = (size_t)blockIdx.x * blockDim.x + threadIdx.x;
  const size_t N = (size_t)B_ * T_ * D_;
  if (i >= N) return;
  size_t d = i % D_;
  size_t r = i / D_;
  size_t t = r % T_;
  size_t b = r / T_;
  xb[((size_t)t * B_ + b) * D_ + d] = f2bf(x[i]);
}

// Pack W_A = [[Wx(all 3U)];[Wh(u,r) | 0]]  (768 x 1536) into WMMA B-operand tiles:
// layout [dir][kt(24)][nt(96)][512 halves]; within a tile, lane L<16 holds column
// nt*16+L with K {0..7,16..23}, lane L+16 same column with K {8..15,24..31}
// (mirror of the documented 16-bit A-matrix striping).
__global__ void k_pack_wa(const float* __restrict__ Wx_f, const float* __restrict__ Wh_f,
                          const float* __restrict__ Wx_b, const float* __restrict__ Wh_b,
                          unsigned short* __restrict__ WA) {
  const size_t PER = (size_t)KT_A * 96 * 512;
  size_t i = (size_t)blockIdx.x * blockDim.x + threadIdx.x;
  if (i >= 2 * PER) return;
  int dir = (int)(i / PER);
  size_t r = i % PER;
  int e = (int)(r & 511);
  size_t tile = r >> 9;
  int nt = (int)(tile % 96);
  int kt = (int)(tile / 96);
  int lane = e >> 4, h = e & 15;
  int n = nt * 16 + (lane & 15);
  int kb = (lane < 16) ? ((h < 8) ? h : h + 8) : ((h < 8) ? h + 8 : h + 16);
  int k = kt * 32 + kb;
  const float* Wx = dir ? Wx_b : Wx_f;
  const float* Wh = dir ? Wh_b : Wh_f;
  float v;
  if (k < D_)            v = Wx[(size_t)k * NG + n];
  else if (n < 2 * U_)   v = Wh[(size_t)(k - D_) * NG + n];
  else                   v = 0.0f;   // candidate recurrent block handled in phase B
  WA[i] = f2bf(v);
}

// Pack Wh_c (512 x 512, columns 1024..1535 of Wh) -> [dir][kt(16)][nt(32)][512]
__global__ void k_pack_wc(const float* __restrict__ Wh_f, const float* __restrict__ Wh_b,
                          unsigned short* __restrict__ WC) {
  const size_t PER = (size_t)KT_B * 32 * 512;
  size_t i = (size_t)blockIdx.x * blockDim.x + threadIdx.x;
  if (i >= 2 * PER) return;
  int dir = (int)(i / PER);
  size_t r = i % PER;
  int e = (int)(r & 511);
  size_t tile = r >> 9;
  int nt = (int)(tile % 32);
  int kt = (int)(tile / 32);
  int lane = e >> 4, h = e & 15;
  int n = 2 * U_ + nt * 16 + (lane & 15);
  int kb = (lane < 16) ? ((h < 8) ? h : h + 8) : ((h < 8) ? h + 8 : h + 16);
  int k = kt * 32 + kb;
  const float* Wh = dir ? Wh_b : Wh_f;
  WC[i] = f2bf(Wh[(size_t)k * NG + n]);
}

// ============================ persistent GRU scan ============================

__global__ void __launch_bounds__(128, 1)
bigru_persistent(const unsigned short* __restrict__ xb,
                 const unsigned short* __restrict__ WA,
                 const unsigned short* __restrict__ WC,
                 const float* __restrict__ bias_f,
                 const float* __restrict__ bias_b,
                 unsigned short* __restrict__ hb,    // [2][B][U] bf16 state
                 float* __restrict__ h32,            // [2][B][U] f32 state
                 float* __restrict__ ubuf,           // [2][B][U] update gate
                 float* __restrict__ cpre,           // [2][B][U] candidate x-part
                 unsigned short* __restrict__ rhb,   // [2][B][U] bf16 r*h
                 float* __restrict__ out,            // [B][T][2U]
                 unsigned* __restrict__ cnt) {
  const int lane  = threadIdx.x & 31;
  const int gw    = blockIdx.x * WAVES_PER_WG + (threadIdx.x >> 5);
  const int col16 = lane & 15;
  const int moff  = (lane >> 4) << 3;   // 0 / 8 : which C rows this lane holds
  const int khoff = (lane >> 4) << 3;   // 0 / 8 : A-operand K sub-offset
  unsigned target = 0;

  // --- static phase-A assignment: one (dir, mtile) row-group per 32 waves ---
  const int groupA = gw >> 5;           // 0..7
  const int wig    = gw & 31;           // 0..31 : n-tile within each gate block
  const int dirA   = groupA >> 2;
  const int mtA    = groupA & 3;
  const int browA  = mtA * 16 + col16;  // A row this lane loads
  // --- static phase-B assignment: exactly one tile per wave ---
  const int dirB = gw >> 7;
  const int mtB  = (gw & 127) >> 5;
  const int ntB  = gw & 31;
  const int browB = mtB * 16 + col16;

  const float* biasA = dirA ? bias_b : bias_f;
  const unsigned short* hrowA = hb + ((size_t)dirA * B_ + browA) * U_;
  const unsigned short* arowB = rhb + ((size_t)dirB * B_ + browB) * U_;

  for (int t = 0; t < T_; ++t) {
    // ------- Phase A: [x_t, h] @ W_A ; 3 n-tiles (u, r, cand) share the A operand -------
    {
      int tx = dirA ? (T_ - 1 - t) : t;                 // backward consumes reversed x
      const unsigned short* xrow = xb + ((size_t)tx * B_ + browA) * D_;

      v8f acc0 = {}, acc1 = {}, acc2 = {};
      v16bf a_cur = load_a(xrow, khoff);                // kt = 0 prefetch
#pragma unroll
      for (int kt = 0; kt < KT_A; ++kt) {
        // prefetch next A tile (double buffer; dead for last iteration)
        v16bf a_nxt = a_cur;
        if (kt + 1 < KT_A) {
          int kn = kt + 1;
          const unsigned short* asrc = (kn < 8) ? xrow : hrowA;
          int k0 = ((kn < 8) ? kn * 32 : (kn - 8) * 32) + khoff;
          a_nxt = load_a(asrc, k0);
        }
        const unsigned short* wp =
            WA + ((((size_t)dirA * KT_A + kt) * 96 + wig) << 9) + (lane << 4);
        v16bf b0 = *(const v16bf*)(wp);                 // u-gate tile   (nt = wig)
        v16bf b1 = *(const v16bf*)(wp + 32 * 512);      // r-gate tile   (nt = wig+32)
        v16bf b2 = *(const v16bf*)(wp + 64 * 512);      // candidate tile(nt = wig+64)
        acc0 = wmma_bf16(a_cur, b0, acc0);              // three independent D->C chains
        acc1 = wmma_bf16(a_cur, b1, acc1);
        acc2 = wmma_bf16(a_cur, b2, acc2);
        a_cur = a_nxt;
      }

      const int col = wig * 16 + col16;                 // column within a U-block
      const size_t base = ((size_t)dirA * B_ + (size_t)(mtA * 16 + moff)) * U_ + col;
      {                                                 // update gate u
        float bv = biasA[col];
#pragma unroll
        for (int v = 0; v < 8; ++v)
          ubuf[base + (size_t)v * U_] = sigm(acc0[v] + bv);
      }
      {                                                 // reset gate r -> r*h (bf16)
        float bv = biasA[U_ + col];
#pragma unroll
        for (int v = 0; v < 8; ++v) {
          float r = sigm(acc1[v] + bv);
          float h = h32[base + (size_t)v * U_];
          rhb[base + (size_t)v * U_] = f2bf(r * h);
        }
      }
      {                                                 // candidate x-part + bias
        float bv = biasA[2 * U_ + col];
#pragma unroll
        for (int v = 0; v < 8; ++v)
          cpre[base + (size_t)v * U_] = acc2[v] + bv;
      }
    }
    grid_sync(cnt, target);

    // ---------------- Phase B: (r*h) @ Wh_c, h update, output ----------------
    {
      v8f accE = {}, accO = {};                         // two chains halve the
      v16bf a_cur = load_a(arowB, khoff);               // exposed XDL accum latency
#pragma unroll
      for (int kt = 0; kt < KT_B; ++kt) {
        v16bf a_nxt = a_cur;
        if (kt + 1 < KT_B) {
          int k0 = (kt + 1) * 32 + khoff;
          a_nxt = load_a(arowB, k0);
        }
        const unsigned short* wp =
            WC + ((((size_t)dirB * KT_B + kt) * 32 + ntB) << 9) + (lane << 4);
        v16bf bm = *(const v16bf*)wp;
        if (kt & 1) accO = wmma_bf16(a_cur, bm, accO);
        else        accE = wmma_bf16(a_cur, bm, accE);
        a_cur = a_nxt;
      }

      int col = ntB * 16 + col16;
      size_t base = ((size_t)dirB * B_ + (size_t)(mtB * 16 + moff)) * U_ + col;
#pragma unroll
      for (int v = 0; v < 8; ++v) {
        size_t idx = base + (size_t)v * U_;
        float hh = ftanh(cpre[idx] + (accE[v] + accO[v]));
        float u  = ubuf[idx];
        float h  = h32[idx];
        float hn = (1.0f - u) * h + u * hh;
        h32[idx] = hn;
        hb[idx]  = f2bf(hn);
        int bi = mtB * 16 + moff + v;
        out[((size_t)bi * T_ + t) * OUTW + (size_t)dirB * U_ + col] = hn;
      }
    }
    grid_sync(cnt, target);
  }
}

// ============================ host launcher ============================

extern "C" void kernel_launch(void* const* d_in, const int* in_sizes, int n_in,
                              void* d_out, int out_size, void* d_ws, size_t ws_size,
                              hipStream_t stream) {
  (void)in_sizes; (void)n_in; (void)out_size; (void)ws_size;

  const float* x    = (const float*)d_in[0];
  const float* Wx_f = (const float*)d_in[1];
  const float* Wh_f = (const float*)d_in[2];
  const float* b_f  = (const float*)d_in[3];
  const float* Wx_b = (const float*)d_in[4];
  const float* Wh_b = (const float*)d_in[5];
  const float* b_b  = (const float*)d_in[6];
  float* out = (float*)d_out;

  // ---- workspace carve-up (all offsets 256B aligned; total ~23.6 MB) ----
  char* p = (char*)d_ws;
  unsigned*       cnt = (unsigned*)p;                              size_t off = 256;
  unsigned short* xb  = (unsigned short*)(p + off);                off += (size_t)2 * T_ * B_ * D_;          // 16 MB
  unsigned short* WA  = (unsigned short*)(p + off);                off += (size_t)2 * 2 * KT_A * 96 * 512;   // 4.7 MB
  unsigned short* WC  = (unsigned short*)(p + off);                off += (size_t)2 * 2 * KT_B * 32 * 512;   // 1.0 MB
  unsigned short* hb  = (unsigned short*)(p + off);                off += (size_t)2 * 2 * B_ * U_;
  float*          h32 = (float*)(p + off);                         off += (size_t)4 * 2 * B_ * U_;
  float*          ub  = (float*)(p + off);                         off += (size_t)4 * 2 * B_ * U_;
  float*          cp  = (float*)(p + off);                         off += (size_t)4 * 2 * B_ * U_;
  unsigned short* rhb = (unsigned short*)(p + off);                off += (size_t)2 * 2 * B_ * U_;

  {
    int n = 2 * B_ * U_;
    k_init<<<(n + 255) / 256, 256, 0, stream>>>(h32, hb, cnt);
  }
  {
    size_t n = (size_t)B_ * T_ * D_;
    k_convert_x<<<(unsigned)((n + 255) / 256), 256, 0, stream>>>(x, xb);
  }
  {
    size_t n = (size_t)2 * KT_A * 96 * 512;
    k_pack_wa<<<(unsigned)((n + 255) / 256), 256, 0, stream>>>(Wx_f, Wh_f, Wx_b, Wh_b, WA);
  }
  {
    size_t n = (size_t)2 * KT_B * 32 * 512;
    k_pack_wc<<<(unsigned)((n + 255) / 256), 256, 0, stream>>>(Wh_f, Wh_b, WC);
  }
  bigru_persistent<<<NWG, 128, 0, stream>>>(xb, WA, WC, b_f, b_b,
                                            hb, h32, ub, cp, rhb, out, cnt);
}
